// ClustCNNMinkEdgeEncoder_17463337026110
// MI455X (gfx1250) — matrix operand order repaired and checked by reading, not compiled, not run on hardware
//
#include <hip/hip_runtime.h>

#define NUM_VOXELS 2000000
#define NUM_CLUSTS 2000
#define NUM_EDGES  16000
#define IN_FEATS   4
#define HIDDEN     128
#define LATENT     64
#define W2T_STRIDE 136   // halves; 272B row stride: bank-conflict-free + 16B aligned

typedef __attribute__((ext_vector_type(16))) _Float16 v16h;
typedef __attribute__((ext_vector_type(8)))  _Float16 v8h;
typedef __attribute__((ext_vector_type(8)))  float    v8f;

// ---------------------------------------------------------------------------
// Kernel 1: zero the cluster accumulator (2000*4 sums + 2000 counts) in d_ws.
// ---------------------------------------------------------------------------
__global__ void zero_ws_kernel(float* __restrict__ ws) {
    int i = blockIdx.x * blockDim.x + threadIdx.x;
    if (i < NUM_CLUSTS * 5) ws[i] = 0.0f;
}

// ---------------------------------------------------------------------------
// Kernel 2: segment-sum of feat = data[:,1:5] and counts, via a 40KB LDS
// histogram per block (ds_add_f32), flushed once with global atomics.
// ws layout: [0 .. 8000) cluster sums (c*4+f), [8000 .. 10000) counts.
// ---------------------------------------------------------------------------
__global__ __launch_bounds__(256) void segsum_kernel(
    const float* __restrict__ data,
    const int*   __restrict__ cid,
    float*       __restrict__ ws)
{
    __shared__ float ls[NUM_CLUSTS * 5];   // 40,000 bytes
    const int tid  = threadIdx.x;
    const int nthr = blockDim.x;

    for (int i = tid; i < NUM_CLUSTS * 5; i += nthr) ls[i] = 0.0f;
    __syncthreads();

    const int stride = gridDim.x * nthr;
    for (int v = blockIdx.x * nthr + tid; v < NUM_VOXELS; v += stride) {
        const int c = cid[v];
        const float* row = data + (size_t)v * 5;
        atomicAdd(&ls[c * 4 + 0], row[1]);
        atomicAdd(&ls[c * 4 + 1], row[2]);
        atomicAdd(&ls[c * 4 + 2], row[3]);
        atomicAdd(&ls[c * 4 + 3], row[4]);
        atomicAdd(&ls[NUM_CLUSTS * 4 + c], 1.0f);
    }
    __syncthreads();

    for (int i = tid; i < NUM_CLUSTS * 5; i += nthr) {
        float s = ls[i];
        if (s != 0.0f) atomicAdd(&ws[i], s);
    }
}

// ---------------------------------------------------------------------------
// Kernel 3: per-edge pooling + 2-layer MLP.
// Each wave32 handles 16 edges; layer 2 (16x64 += 16x128 @ 128x64) runs on
// v_wmma_f32_16x16x32_f16 (4 N-tiles x 4 K-steps). W2 is held in LDS
// transposed (N-major, padded stride) so each B fragment is two ds_load_b128.
// ---------------------------------------------------------------------------
__global__ __launch_bounds__(128) void edge_mlp_kernel(
    const int*   __restrict__ edge_index,   // (2, NUM_EDGES)
    const float* __restrict__ W1,           // (4, 128)
    const float* __restrict__ b1,           // (128)
    const float* __restrict__ W2,           // (128, 64)
    const float* __restrict__ b2,           // (64)
    const float* __restrict__ ws,           // cluster sums + counts
    float*       __restrict__ out)          // (NUM_EDGES, 64)
{
    __shared__ __align__(16) _Float16 sW2t[LATENT * W2T_STRIDE]; // ~17 KB
    __shared__ float sW1[IN_FEATS * HIDDEN];                     // 2 KB
    __shared__ float sB1[HIDDEN];

    const int tid = threadIdx.x;
    // W2 (k,n) row-major in global -> transposed sW2t[n*stride + k]
    for (int i = tid; i < HIDDEN * LATENT; i += 128) {
        const int k = i >> 6;           // i / LATENT
        const int n = i & (LATENT - 1); // i % LATENT
        sW2t[n * W2T_STRIDE + k] = (_Float16)W2[i];
    }
    for (int i = tid; i < IN_FEATS * HIDDEN; i += 128) sW1[i] = W1[i];
    if (tid < HIDDEN) sB1[tid] = b1[tid];
    __syncthreads();

    const int lane  = tid & 31;
    const int wave  = tid >> 5;
    const int r     = lane & 15;     // row in edge tile / column in N-tile
    const int hi    = lane >> 4;     // which half of the wave
    const int ebase = (blockIdx.x * 4 + wave) * 16;
    const int e     = ebase + r;

    // ---- pooled features for edge row r (both wave halves duplicate) ----
    const int e0 = edge_index[e];
    const int e1 = edge_index[NUM_EDGES + e];
    const float* cs = ws;
    const float* cc = ws + NUM_CLUSTS * 4;
    const float inv = 1.0f / fmaxf(cc[e0] + cc[e1], 1.0f);
    const float p0 = (cs[e0 * 4 + 0] + cs[e1 * 4 + 0]) * inv;
    const float p1 = (cs[e0 * 4 + 1] + cs[e1 * 4 + 1]) * inv;
    const float p2 = (cs[e0 * 4 + 2] + cs[e1 * 4 + 2]) * inv;
    const float p3 = (cs[e0 * 4 + 3] + cs[e1 * 4 + 3]) * inv;

    v8f acc[4];
    #pragma unroll
    for (int nt = 0; nt < 4; ++nt) acc[nt] = (v8f){};

    #pragma unroll
    for (int step = 0; step < 4; ++step) {
        // ---- A fragment: h = relu(pooled @ W1 + b1), computed straight into
        //      the 16-bit 16x32 A layout (lane holds row r). ----
        v16h a;
        #pragma unroll
        for (int i = 0; i < 16; ++i) {
            const int K = step * 32 + hi * 16 + i;
            float hv = sB1[K]
                     + p0 * sW1[0 * HIDDEN + K]
                     + p1 * sW1[1 * HIDDEN + K]
                     + p2 * sW1[2 * HIDDEN + K]
                     + p3 * sW1[3 * HIDDEN + K];
            a[i] = (_Float16)fmaxf(hv, 0.0f);
        }

        // ---- B fragments: lane holds column col; halves K = step*32+hi*16+i
        //      are contiguous in the transposed LDS image -> 2x ds_load_b128. ----
        #pragma unroll
        for (int nt = 0; nt < 4; ++nt) {
            const int col = nt * 16 + r;
            const int off = col * W2T_STRIDE + step * 32 + hi * 16;
            union { v16h v; v8h h[2]; } bu;
            bu.h[0] = *(const v8h*)&sW2t[off];
            bu.h[1] = *(const v8h*)&sW2t[off + 8];
            acc[nt] = __builtin_amdgcn_wmma_f32_16x16x32_f16(
                false, a, false, bu.v, (short)0, acc[nt], false, false);
        }
    }

    // ---- store: D layout lane = N (=r), VGPR v -> M = v + 8*hi ----
    #pragma unroll
    for (int nt = 0; nt < 4; ++nt) {
        const int   col  = nt * 16 + r;
        const float bias = b2[col];
        #pragma unroll
        for (int v = 0; v < 8; ++v) {
            const int row = v + hi * 8;
            out[(size_t)(ebase + row) * LATENT + col] = acc[nt][v] + bias;
        }
    }
}

// ---------------------------------------------------------------------------
extern "C" void kernel_launch(void* const* d_in, const int* in_sizes, int n_in,
                              void* d_out, int out_size, void* d_ws, size_t ws_size,
                              hipStream_t stream) {
    const float* data = (const float*)d_in[0];
    const int*   cid  = (const int*)  d_in[1];
    const int*   ei   = (const int*)  d_in[2];
    const float* W1   = (const float*)d_in[3];
    const float* b1   = (const float*)d_in[4];
    const float* W2   = (const float*)d_in[5];
    const float* b2   = (const float*)d_in[6];
    float* out = (float*)d_out;
    float* ws  = (float*)d_ws;

    zero_ws_kernel<<<(NUM_CLUSTS * 5 + 255) / 256, 256, 0, stream>>>(ws);
    segsum_kernel<<<128, 256, 0, stream>>>(data, cid, ws);
    edge_mlp_kernel<<<NUM_EDGES / 64, 128, 0, stream>>>(ei, W1, b1, W2, b2, ws, out);
}
